// RankAttentionDeblureNet_83915071029458
// MI455X (gfx1250) — matrix operand order repaired and checked by reading, not compiled, not run on hardware
//
#include <hip/hip_runtime.h>
#include <stddef.h>

// ---------------------------------------------------------------------------
// Types / constants
// ---------------------------------------------------------------------------
typedef __attribute__((ext_vector_type(16))) __bf16 v16bf;
typedef __attribute__((ext_vector_type(8)))  float  v8f;

// async-load-to-LDS builtin wants int4-vector pointers in AS1 (global) / AS3 (LDS)
typedef int v4i __attribute__((vector_size(16)));
typedef __attribute__((address_space(1))) v4i gv4i;
typedef __attribute__((address_space(3))) v4i lv4i;

#if __has_builtin(__builtin_amdgcn_global_load_async_to_lds_b128) && \
    __has_builtin(__builtin_amdgcn_s_wait_asynccnt)
#define USE_ASYNC_LDS 1
#else
#define USE_ASYNC_LDS 0
#endif

enum { GBM = 64, GBN = 256, GBK = 32, LDT = GBK + 8 };   // GEMM tiling
enum { F_RELU = 1, F_TANH = 2, F_RESID = 4 };
#define NB 2   // batch size (reference: B=2, H=W=256)

union Frag { v16bf v; uint4 q[2]; };

// ---------------------------------------------------------------------------
// GEMM: C[M,N] = A[M,K] * B[N,K]^T,  A/B bf16 row-major, C f32.
// Fused epilogue: per-row scale/shift (BN or bias), residual add, relu, tanh.
// wave32: 8 waves in 2x4 grid; each wave 32x64 via 2x4 v_wmma 16x16x32 bf16.
// Tiles staged via GLOBAL_LOAD_ASYNC_TO_LDS (ASYNCcnt) when available.
// Requires K % 32 == 0 and N % 256 == 0 (producers guarantee via padding).
// ---------------------------------------------------------------------------
__global__ __launch_bounds__(256) void gemm_bf16_k(
    const __bf16* __restrict__ A, const __bf16* __restrict__ Bm,
    float* __restrict__ C, int M, int N, int K,
    const float* __restrict__ scale, const float* __restrict__ shift,
    const float* __restrict__ resid, int flags)
{
  __shared__ __align__(16) __bf16 As[GBM * LDT];
  __shared__ __align__(16) __bf16 Bs[GBN * LDT];

  const int tid  = threadIdx.x;
  const int lane = tid & 31;
  const int wave = tid >> 5;
  const int wm   = (wave >> 2) * 32;   // wave M offset in tile (0/32)
  const int wn   = (wave & 3)  * 64;   // wave N offset in tile (0..192)
  const int lrow = tid >> 2;           // 0..63  (staging row)
  const int lkc  = (tid & 3) * 8;      // 0,8,16,24 (staging k-chunk, halves)
  const int lm   = lane & 15;
  const int hi   = lane >> 4;

  v8f zero8 = {0.f,0.f,0.f,0.f,0.f,0.f,0.f,0.f};
  v8f acc[2][4];
#pragma unroll
  for (int mt = 0; mt < 2; ++mt)
#pragma unroll
    for (int nt = 0; nt < 4; ++nt) acc[mt][nt] = zero8;

  for (int kt = 0; kt < K; kt += GBK) {
    // ---- stage A (64x32) and B (256x32) tiles into LDS ---------------------
    int gr = blockIdx.y * GBM + lrow;
#if USE_ASYNC_LDS
    {
      // rows >= M only feed masked output rows; clamp address instead of zero
      int grc = gr < M ? gr : (M - 1);
      __builtin_amdgcn_global_load_async_to_lds_b128(
          (gv4i*)(A + (size_t)grc * K + kt + lkc),
          (lv4i*)&As[lrow * LDT + lkc], 0, 0);
    }
#pragma unroll
    for (int h = 0; h < 4; ++h) {
      int gc = blockIdx.x * GBN + lrow + 64 * h;
      __builtin_amdgcn_global_load_async_to_lds_b128(
          (gv4i*)(Bm + (size_t)gc * K + kt + lkc),
          (lv4i*)&Bs[(lrow + 64 * h) * LDT + lkc], 0, 0);
    }
    if (kt + GBK < K)   // lowers to global_prefetch_b8
      __builtin_prefetch(A + (size_t)(gr < M ? gr : 0) * K + kt + GBK + lkc, 0, 3);
    __builtin_amdgcn_s_wait_asynccnt(0);
#else
    {
      uint4 av = {0u, 0u, 0u, 0u};
      if (gr < M) av = *(const uint4*)(A + (size_t)gr * K + kt + lkc);
      *(uint4*)&As[lrow * LDT + lkc] = av;
#pragma unroll
      for (int h = 0; h < 4; ++h) {
        int gc = blockIdx.x * GBN + lrow + 64 * h;
        *(uint4*)&Bs[(lrow + 64 * h) * LDT + lkc] =
            *(const uint4*)(Bm + (size_t)gc * K + kt + lkc);
      }
      if (kt + GBK < K) {
        if (gr < M) __builtin_prefetch(A + (size_t)gr * K + kt + GBK + lkc, 0, 3);
        __builtin_prefetch(Bm + (size_t)(blockIdx.x * GBN + lrow) * K + kt + GBK + lkc, 0, 3);
      }
    }
#endif
    __syncthreads();

    // ---- fragments per CDNA5 WMMA VGPR layouts -----------------------------
    Frag fa[2], fb[4];
#pragma unroll
    for (int mt = 0; mt < 2; ++mt) {
      // A 16x32 bf16: lane<16 holds K {0..7,16..23}; lane>=16 {8..15,24..31}
      const __bf16* p = &As[(wm + mt * 16 + lm) * LDT + hi * 8];
      fa[mt].q[0] = *(const uint4*)p;
      fa[mt].q[1] = *(const uint4*)(p + 16);
    }
#pragma unroll
    for (int nt = 0; nt < 4; ++nt) {
      // B 32x16 bf16: lane<16 -> col n, K {0..7,16..23}; lane>=16 -> col n, K {8..15,24..31}
      const __bf16* p = &Bs[(wn + nt * 16 + lm) * LDT + hi * 8];
      fb[nt].q[0] = *(const uint4*)p;
      fb[nt].q[1] = *(const uint4*)(p + 16);
    }

#pragma unroll
    for (int mt = 0; mt < 2; ++mt)
#pragma unroll
      for (int nt = 0; nt < 4; ++nt)
        acc[mt][nt] = __builtin_amdgcn_wmma_f32_16x16x32_bf16(
            false, fa[mt].v, false, fb[nt].v, (short)0, acc[mt][nt],
            false, false);
    __syncthreads();
  }

  // ---- fused epilogue; C/D layout: VGPR r -> M = r + 8*hi, lane&15 -> N ----
#pragma unroll
  for (int mt = 0; mt < 2; ++mt) {
#pragma unroll
    for (int nt = 0; nt < 4; ++nt) {
#pragma unroll
      for (int r = 0; r < 8; ++r) {
        int m = blockIdx.y * GBM + wm + mt * 16 + hi * 8 + r;
        int n = blockIdx.x * GBN + wn + nt * 16 + lm;
        if (m < M) {
          float v = acc[mt][nt][r];
          if (scale) v *= scale[m];
          if (shift) v += shift[m];
          if (flags & F_RESID) v += resid[(size_t)m * N + n];
          if (flags & F_RELU)  v = fmaxf(v, 0.f);
          if (flags & F_TANH)  v = tanhf(v);
          C[(size_t)m * N + n] = v;
        }
      }
    }
  }
}

// ---------------------------------------------------------------------------
// im2col (f32 NCHW slice -> bf16 [L][k_total] patch matrix), supports
// stride / pad / input (lhs) dilation, concat offset and K zero-padding.
// ---------------------------------------------------------------------------
__global__ void im2col_k(const float* __restrict__ src, __bf16* __restrict__ dst,
                         int C, int H, int W, int OH, int OW, int KH, int KW,
                         int stride, int pad, int ldil,
                         int k_base, int k_total, int region_k)
{
  long long t = (long long)blockIdx.x * 256 + threadIdx.x;
  long long L = (long long)OH * OW;
  if (t >= L * region_k) return;
  int  kk = (int)(t % region_k);
  long long l = t / region_k;
  int Kreal = C * KH * KW;
  float val = 0.f;
  if (kk < Kreal) {
    int c = kk / (KH * KW), r = kk % (KH * KW);
    int i = r / KW, j = r % KW;
    int oy = (int)(l / OW), ox = (int)(l % OW);
    int vy = oy * stride - pad + i;
    int vx = ox * stride - pad + j;
    if (vy >= 0 && vx >= 0 && (vy % ldil) == 0 && (vx % ldil) == 0) {
      int y = vy / ldil, x = vx / ldil;
      if (y < H && x < W) val = src[((size_t)c * H + y) * W + x];
    }
  }
  dst[l * k_total + k_base + kk] = (__bf16)val;
}

// f32 OIHW weights -> bf16 [OC][Kp] (zero-padded K)
__global__ void wcvt_k(const float* __restrict__ w, __bf16* __restrict__ out,
                       int OC, int Kreal, int Kp)
{
  int t = blockIdx.x * 256 + threadIdx.x;
  if (t >= OC * Kp) return;
  int oc = t / Kp, k = t % Kp;
  out[t] = (__bf16)(k < Kreal ? w[(size_t)oc * Kreal + k] : 0.f);
}

// ConvTranspose weights (IC,OC,3,3) -> flipped/swapped bf16 [OC][IC*9]
__global__ void wtcvt_k(const float* __restrict__ w, __bf16* __restrict__ out,
                        int IC, int OC)
{
  int t = blockIdx.x * 256 + threadIdx.x;
  if (t >= OC * IC * 9) return;
  int oc = t / (IC * 9), r = t % (IC * 9);
  int ic = r / 9, s = r % 9, i = s / 3, j = s % 3;
  out[((size_t)oc * IC + ic) * 9 + i * 3 + j] =
      (__bf16)w[(((size_t)ic * OC + oc) * 3 + (2 - i)) * 3 + (2 - j)];
}

// BN folding: scale = g*rsqrt(v+eps), shift = b - m*scale
__global__ void bnprep_k(const float* __restrict__ g, const float* __restrict__ b,
                         const float* __restrict__ m, const float* __restrict__ v,
                         float* __restrict__ scale, float* __restrict__ shift, int C)
{
  int c = blockIdx.x * 256 + threadIdx.x;
  if (c >= C) return;
  float s = g[c] * rsqrtf(v[c] + 1e-5f);
  scale[c] = s;
  shift[c] = b[c] - m[c] * s;
}

// unfold(3x3,pad1) + per-patch L2 normalization -> bf16 [L][C*9]
// one wave per output pixel; wave32 shuffle reduction
__global__ void unfold_norm_k(const float* __restrict__ f, __bf16* __restrict__ U,
                              int C, int H, int W)
{
  int wave = threadIdx.x >> 5, lane = threadIdx.x & 31;
  int l = blockIdx.x * 8 + wave;
  int L = H * W;
  if (l >= L) return;
  int y0 = l / W, x0 = l % W, C9 = C * 9;
  float ss = 0.f;
  for (int k = lane; k < C9; k += 32) {
    int c = k / 9, s = k % 9, i = s / 3, j = s % 3;
    int y = y0 + i - 1, x = x0 + j - 1;
    float v = (y >= 0 && y < H && x >= 0 && x < W)
                  ? f[((size_t)c * H + y) * W + x] : 0.f;
    ss += v * v;
  }
  for (int o = 16; o; o >>= 1) ss += __shfl_xor(ss, o, 32);
  float inv = 1.f / fmaxf(sqrtf(ss), 1e-12f);
  for (int k = lane; k < C9; k += 32) {
    int c = k / 9, s = k % 9, i = s / 3, j = s % 3;
    int y = y0 + i - 1, x = x0 + j - 1;
    float v = (y >= 0 && y < H && x >= 0 && x < W)
                  ? f[((size_t)c * H + y) * W + x] : 0.f;
    U[(size_t)l * C9 + k] = (__bf16)(v * inv);
  }
}

// hard 1-NN: idx[lc] = argmax_lr rel[lr][lc]  (coalesced across columns)
__global__ void colargmax_k(const float* __restrict__ rel, int* __restrict__ idx, int L)
{
  int lc = blockIdx.x * 256 + threadIdx.x;
  if (lc >= L) return;
  float best = -3.4e38f; int bi = 0;
  for (int lr = 0; lr < L; ++lr) {
    float v = rel[(size_t)lr * L + lc];
    if (v > best) { best = v; bi = lr; }
  }
  idx[lc] = bi;
}

// gather ref patches by idx, fold(3x3,pad1), add f_cur  (single batch slice)
__global__ void gather_fold_k(const float* __restrict__ f_ref,
                              const float* __restrict__ f_cur,
                              const int* __restrict__ idx,
                              float* __restrict__ out, int C, int H, int W)
{
  long long t = (long long)blockIdx.x * 256 + threadIdx.x;
  if (t >= (long long)C * H * W) return;
  int c = (int)(t / (H * W)), r = (int)(t % (H * W));
  int y = r / W, x = r % W;
  float acc = f_cur[t];
  for (int i = 0; i < 3; ++i)
    for (int j = 0; j < 3; ++j) {
      int h = y + 1 - i, w = x + 1 - j;
      if (h < 0 || h >= H || w < 0 || w >= W) continue;
      int l2 = idx[h * W + w];
      int ry = l2 / W + i - 1, rx = l2 % W + j - 1;
      if (ry >= 0 && ry < H && rx >= 0 && rx < W)
        acc += f_ref[((size_t)c * H + ry) * W + rx];
    }
  out[t] = acc;
}

// ---------------------------------------------------------------------------
// Host-side orchestration
// ---------------------------------------------------------------------------
struct BNp { const float *b, *g, *m, *v; };
struct Blk { BNp bn1, bn2; const float *c1, *c2; int down; BNp dbn; const float* dw; };
struct SS  { float* scale; float* shift; };
struct ConvW { const __bf16* w; int OC, C, KH, KW, Kp; };
struct Enc { float *e1, *e2, *e3, *e4; };

struct Ws {
  char* base; size_t off, cap;
  void* alloc(size_t n) {
    size_t a = (off + 255) & ~(size_t)255;
    off = a + n;
    return base + a;
  }
  void reset(size_t mark) { off = mark; }
};

extern "C" void kernel_launch(void* const* d_in, const int* in_sizes, int n_in,
                              void* d_out, int out_size, void* d_ws, size_t ws_size,
                              hipStream_t stream)
{
  (void)in_sizes; (void)n_in; (void)out_size;
  const float* input_current = (const float*)d_in[0];
  const float* real_current  = (const float*)d_in[1];
  const float* real_blur     = (const float*)d_in[2];

  // ---- params: JAX pytree flattening order (dict keys sorted) --------------
  int pi = 3;
  auto nxt = [&]() -> const float* { return (const float*)d_in[pi++]; };
  const float *cc1_b = nxt(), *cc1_w = nxt(), *cc2_b = nxt(), *cc2_w = nxt(),
              *cc3_b = nxt(), *cc3_w = nxt(), *cc4_b = nxt(), *cc4_w = nxt();
  const float *conv1_w = nxt();
  const float *f128_b = nxt(), *f128_w = nxt(), *f256_b = nxt(), *f256_w = nxt(),
              *f512_b = nxt(), *f512_w = nxt(), *f64_b = nxt(),  *f64_w = nxt();
  const float *last_b = nxt(), *last_w = nxt();

  auto rdBN  = [&](BNp& p) { p.b = nxt(); p.g = nxt(); p.m = nxt(); p.v = nxt(); };
  auto rdBlk = [&](Blk& k, bool down) {
    rdBN(k.bn1); rdBN(k.bn2); k.c1 = nxt(); k.c2 = nxt();
    k.down = down ? 1 : 0;
    if (down) { rdBN(k.dbn); k.dw = nxt(); } else { k.dw = nullptr; }
  };
  Blk L1[3], L2[3], L3[3], L4[3], L5[3], L6[3], L7[3], L8[3];
  auto rdLayer = [&](Blk* L, bool d0) { rdBlk(L[0], d0); rdBlk(L[1], false); rdBlk(L[2], false); };
  rdLayer(L1, false); rdLayer(L2, true); rdLayer(L3, true); rdLayer(L4, true);
  rdLayer(L5, false); rdLayer(L6, false); rdLayer(L7, false); rdLayer(L8, false);
  const float *up0_b = nxt(), *up0_w = nxt(), *up1_b = nxt(), *up1_w = nxt(),
              *up2_b = nxt(), *up2_w = nxt(), *up3_b = nxt(), *up3_w = nxt();

  Ws ws{(char*)d_ws, 0, ws_size};

  // ---- device-side conversions --------------------------------------------
  auto cvt_w = [&](const float* w, int OC, int C, int KH, int KW) -> ConvW {
    int Kreal = C * KH * KW, Kp = (Kreal + 31) & ~31;
    __bf16* dw = (__bf16*)ws.alloc((size_t)OC * Kp * sizeof(__bf16));
    int n = OC * Kp;
    wcvt_k<<<(n + 255) / 256, 256, 0, stream>>>(w, dw, OC, Kreal, Kp);
    ConvW c{dw, OC, C, KH, KW, Kp};
    return c;
  };
  auto cvt_wt = [&](const float* w, int IC, int OC) -> ConvW {
    int Kp = IC * 9;  // IC multiple of 32 -> Kp % 32 == 0
    __bf16* dw = (__bf16*)ws.alloc((size_t)OC * Kp * sizeof(__bf16));
    int n = OC * IC * 9;
    wtcvt_k<<<(n + 255) / 256, 256, 0, stream>>>(w, dw, IC, OC);
    ConvW c{dw, OC, IC, 3, 3, Kp};
    return c;
  };
  auto prep_bn = [&](const BNp& p, int C) -> SS {
    SS s;
    s.scale = (float*)ws.alloc((size_t)C * 4);
    s.shift = (float*)ws.alloc((size_t)C * 4);
    bnprep_k<<<(C + 255) / 256, 256, 0, stream>>>(p.g, p.b, p.m, p.v,
                                                  s.scale, s.shift, C);
    return s;
  };

  // ---- conv executor: im2col + WMMA GEMM per batch -------------------------
  auto conv_run = [&](const ConvW& cw, const float* src, float* dst,
                      int H, int W, int OH, int OW, int stride, int pad, int ldil,
                      const float* scale, const float* shift,
                      const float* resid, int flags) {
    long long L = (long long)OH * OW;
    size_t mark = ws.off;
    __bf16* Bm = (__bf16*)ws.alloc((size_t)L * cw.Kp * sizeof(__bf16));
    for (int b = 0; b < NB; ++b) {
      long long n = L * cw.Kp;
      im2col_k<<<(unsigned)((n + 255) / 256), 256, 0, stream>>>(
          src + (size_t)b * cw.C * H * W, Bm, cw.C, H, W, OH, OW, cw.KH, cw.KW,
          stride, pad, ldil, 0, cw.Kp, cw.Kp);
      dim3 g((unsigned)(L / GBN), (unsigned)((cw.OC + GBM - 1) / GBM));
      gemm_bf16_k<<<g, 256, 0, stream>>>(
          cw.w, Bm, dst + (size_t)b * cw.OC * L, cw.OC, (int)L, cw.Kp,
          scale, shift, resid ? resid + (size_t)b * cw.OC * L : nullptr, flags);
    }
    ws.reset(mark);
  };

  // concat (up to 3 sources) -> 3x3 pad1 stride1 conv
  auto conv_cat = [&](const ConvW& cw, const float* s1, int C1,
                      const float* s2, int C2, const float* s3, int C3,
                      float* dst, int H, int W, const float* bias, int flags) {
    long long L = (long long)H * W;
    size_t mark = ws.off;
    __bf16* Bm = (__bf16*)ws.alloc((size_t)L * cw.Kp * sizeof(__bf16));
    for (int b = 0; b < NB; ++b) {
      int kb = 0;
      auto part = [&](const float* s, int Cc) {
        if (!s || !Cc) return;
        long long n = L * (long long)Cc * 9;
        im2col_k<<<(unsigned)((n + 255) / 256), 256, 0, stream>>>(
            s + (size_t)b * Cc * H * W, Bm, Cc, H, W, H, W, 3, 3,
            1, 1, 1, kb, cw.Kp, Cc * 9);
        kb += Cc * 9;
      };
      part(s1, C1); part(s2, C2); part(s3, C3);
      dim3 g((unsigned)(L / GBN), (unsigned)((cw.OC + GBM - 1) / GBM));
      gemm_bf16_k<<<g, 256, 0, stream>>>(
          cw.w, Bm, dst + (size_t)b * cw.OC * L, cw.OC, (int)L, cw.Kp,
          nullptr, bias, nullptr, flags);
    }
    ws.reset(mark);
  };

  // ResNet basic block (BN folded into GEMM epilogue; residual fused)
  auto basic_block = [&](const Blk& k, const float* x, float* out,
                         int C, int H, int W, int planes) {
    int stride = k.down ? 2 : 1;
    int OH = H / stride, OW = W / stride;
    size_t mark = ws.off;
    float* tmp = (float*)ws.alloc((size_t)NB * planes * OH * OW * 4);
    SS b1 = prep_bn(k.bn1, planes), b2 = prep_bn(k.bn2, planes);
    ConvW c1 = cvt_w(k.c1, planes, C, 3, 3);
    conv_run(c1, x, tmp, H, W, OH, OW, stride, 1, 1,
             b1.scale, b1.shift, nullptr, F_RELU);
    const float* res = x;
    if (k.down) {
      SS db = prep_bn(k.dbn, planes);
      ConvW dc = cvt_w(k.dw, planes, C, 1, 1);
      float* sc = (float*)ws.alloc((size_t)NB * planes * OH * OW * 4);
      conv_run(dc, x, sc, H, W, OH, OW, stride, 0, 1,
               db.scale, db.shift, nullptr, 0);
      res = sc;
    }
    ConvW c2 = cvt_w(k.c2, planes, planes, 3, 3);
    conv_run(c2, tmp, out, OH, OW, OH, OW, 1, 1, 1,
             b2.scale, b2.shift, res, F_RELU | F_RESID);
    ws.reset(mark);
  };

  auto run_layer = [&](const Blk* Lb, int n, const float* x, float* out,
                       int Cin, int planes, int H, int W) {
    int s0 = Lb[0].down ? 2 : 1;
    int OH = H / s0, OW = W / s0;
    size_t mark = ws.off;
    const float* cur = x; int curC = Cin, curH = H, curW = W;
    for (int i = 0; i < n; ++i) {
      float* dst = (i == n - 1) ? out
                                : (float*)ws.alloc((size_t)NB * planes * OH * OW * 4);
      basic_block(Lb[i], cur, dst, curC, curH, curW, planes);
      cur = dst; curC = planes; curH = OH; curW = OW;
    }
    ws.reset(mark);
  };

  // encoder (conv1 7x7 s2 + relu, then layers 1..4)
  auto encode = [&](const float* x) -> Enc {
    Enc E;
    E.e1 = (float*)ws.alloc(2ull * 64 * 128 * 128 * 4);
    {
      size_t mk = ws.off;
      float* t0 = (float*)ws.alloc(2ull * 64 * 128 * 128 * 4);
      ConvW c = cvt_w(conv1_w, 64, 3, 7, 7);
      conv_run(c, x, t0, 256, 256, 128, 128, 2, 3, 1,
               nullptr, nullptr, nullptr, F_RELU);
      run_layer(L1, 3, t0, E.e1, 64, 64, 128, 128);
      ws.reset(mk);
    }
    E.e2 = (float*)ws.alloc(2ull * 128 * 64 * 64 * 4);
    run_layer(L2, 3, E.e1, E.e2, 64, 128, 128, 128);
    E.e3 = (float*)ws.alloc(2ull * 256 * 32 * 32 * 4);
    run_layer(L3, 3, E.e2, E.e3, 128, 256, 64, 64);
    E.e4 = (float*)ws.alloc(2ull * 512 * 16 * 16 * 4);
    run_layer(L4, 3, E.e3, E.e4, 256, 512, 32, 32);
    return E;
  };

  // ranking correlation: unfold+L2norm -> WMMA GEMM -> argmax -> gather+fold
  auto correlate = [&](const float* f_cur, const float* f_ref, const float* f_blur,
                       float* out, int C, int H, int W) {
    int L = H * W, C9 = C * 9;
    size_t mark = ws.off;
    __bf16* Ucur  = (__bf16*)ws.alloc((size_t)L * C9 * sizeof(__bf16));
    __bf16* Ublur = (__bf16*)ws.alloc((size_t)L * C9 * sizeof(__bf16));
    float*  rel   = (float*)ws.alloc((size_t)L * L * 4);
    int*    idx   = (int*)ws.alloc((size_t)L * 4);
    for (int b = 0; b < NB; ++b) {
      const float* fc = f_cur + (size_t)b * C * L;
      const float* fr = f_ref + (size_t)b * C * L;
      const float* fb = f_blur + (size_t)b * C * L;
      unfold_norm_k<<<L / 8, 256, 0, stream>>>(fc, Ucur, C, H, W);
      unfold_norm_k<<<L / 8, 256, 0, stream>>>(fb, Ublur, C, H, W);
      dim3 g((unsigned)(L / GBN), (unsigned)((L + GBM - 1) / GBM));
      gemm_bf16_k<<<g, 256, 0, stream>>>(Ublur, Ucur, rel, L, L, C9,
                                         nullptr, nullptr, nullptr, 0);
      colargmax_k<<<(L + 255) / 256, 256, 0, stream>>>(rel, idx, L);
      long long n = (long long)C * L;
      gather_fold_k<<<(unsigned)((n + 255) / 256), 256, 0, stream>>>(
          fr, fc, idx, out + (size_t)b * C * L, C, H, W);
    }
    ws.reset(mark);
  };

  // =========================== forward pass ================================
  Enc I = encode(input_current);   // iE1..iE3, f_cur = I.e4
  Enc R = encode(real_blur);       // rE*b,     f_ref_blur = R.e4

  // feature pyramid on real_current (bias + relu, 5x5 s2 p2)
  float* rE1 = (float*)ws.alloc(2ull * 64 * 128 * 128 * 4);
  { ConvW c = cvt_w(f64_w, 64, 3, 5, 5);
    conv_run(c, real_current, rE1, 256, 256, 128, 128, 2, 2, 1,
             nullptr, f64_b, nullptr, F_RELU); }
  float* rE2 = (float*)ws.alloc(2ull * 128 * 64 * 64 * 4);
  { ConvW c = cvt_w(f128_w, 128, 64, 5, 5);
    conv_run(c, rE1, rE2, 128, 128, 64, 64, 2, 2, 1,
             nullptr, f128_b, nullptr, F_RELU); }
  float* rE3 = (float*)ws.alloc(2ull * 256 * 32 * 32 * 4);
  { ConvW c = cvt_w(f256_w, 256, 128, 5, 5);
    conv_run(c, rE2, rE3, 64, 64, 32, 32, 2, 2, 1,
             nullptr, f256_b, nullptr, F_RELU); }
  float* f_ref = (float*)ws.alloc(2ull * 512 * 16 * 16 * 4);
  { ConvW c = cvt_w(f512_w, 512, 256, 5, 5);
    conv_run(c, rE3, f_ref, 32, 32, 16, 16, 2, 2, 1,
             nullptr, f512_b, nullptr, F_RELU); }

  // ranking correlations
  float* r4 = (float*)ws.alloc(2ull * 512 * 256 * 4);
  correlate(I.e4, f_ref, R.e4, r4, 512, 16, 16);
  float* r3 = (float*)ws.alloc(2ull * 256 * 1024 * 4);
  correlate(I.e3, rE3, R.e3, r3, 256, 32, 32);
  float* r2 = (float*)ws.alloc(2ull * 128 * 4096 * 4);
  correlate(I.e2, rE2, R.e2, r2, 128, 64, 64);

  // decoder
  float* d4 = (float*)ws.alloc(2ull * 512 * 256 * 4);
  { ConvW c = cvt_w(cc4_w, 512, 1024, 3, 3);
    conv_cat(c, r4, 512, I.e4, 512, nullptr, 0, d4, 16, 16, cc4_b, 0); }
  float* d4o = (float*)ws.alloc(2ull * 512 * 256 * 4);
  run_layer(L5, 3, d4, d4o, 512, 512, 16, 16);

  float* d3u = (float*)ws.alloc(2ull * 256 * 1024 * 4);
  { ConvW c = cvt_wt(up3_w, 512, 256);
    conv_run(c, d4o, d3u, 16, 16, 32, 32, 1, 1, 2, nullptr, up3_b, nullptr, 0); }
  float* d3 = (float*)ws.alloc(2ull * 256 * 1024 * 4);
  { ConvW c = cvt_w(cc3_w, 256, 768, 3, 3);
    conv_cat(c, I.e3, 256, d3u, 256, r3, 256, d3, 32, 32, cc3_b, 0); }
  float* d3o = (float*)ws.alloc(2ull * 256 * 1024 * 4);
  run_layer(L6, 3, d3, d3o, 256, 256, 32, 32);

  float* d2u = (float*)ws.alloc(2ull * 128 * 4096 * 4);
  { ConvW c = cvt_wt(up2_w, 256, 128);
    conv_run(c, d3o, d2u, 32, 32, 64, 64, 1, 1, 2, nullptr, up2_b, nullptr, 0); }
  float* d2 = (float*)ws.alloc(2ull * 128 * 4096 * 4);
  { ConvW c = cvt_w(cc2_w, 128, 384, 3, 3);
    conv_cat(c, I.e2, 128, d2u, 128, r2, 128, d2, 64, 64, cc2_b, 0); }
  float* d2o = (float*)ws.alloc(2ull * 128 * 4096 * 4);
  run_layer(L7, 3, d2, d2o, 128, 128, 64, 64);

  float* d1u = (float*)ws.alloc(2ull * 64 * 16384 * 4);
  { ConvW c = cvt_wt(up1_w, 128, 64);
    conv_run(c, d2o, d1u, 64, 64, 128, 128, 1, 1, 2, nullptr, up1_b, nullptr, 0); }
  float* d1 = (float*)ws.alloc(2ull * 64 * 16384 * 4);
  { ConvW c = cvt_w(cc1_w, 64, 128, 3, 3);
    conv_cat(c, I.e1, 64, d1u, 64, nullptr, 0, d1, 128, 128, cc1_b, 0); }
  float* d1o = (float*)ws.alloc(2ull * 64 * 16384 * 4);
  run_layer(L8, 3, d1, d1o, 64, 64, 128, 128);

  float* u0 = (float*)ws.alloc(2ull * 32 * 65536 * 4);
  { ConvW c = cvt_wt(up0_w, 64, 32);
    conv_run(c, d1o, u0, 128, 128, 256, 256, 1, 1, 2, nullptr, up0_b, nullptr, 0); }
  { ConvW c = cvt_w(last_w, 3, 32, 1, 1);  // 1x1 conv + bias + tanh -> d_out
    conv_run(c, u0, (float*)d_out, 256, 256, 256, 256, 1, 0, 1,
             nullptr, last_b, nullptr, F_TANH); }
}
